// Head_30709016166761
// MI455X (gfx1250) — matrix operand order repaired and checked by reading, not compiled, not run on hardware
//
#include <hip/hip_runtime.h>
#include <hip/hip_bf16.h>

// ---------------------------------------------------------------------------
// Causal single-head attention, B=8 T=2048 C=768 H=64, fp32 in/out.
// Pipeline: (1) QKV projection GEMMs -> bf16 workspace, (2) flash attention.
// All matmuls on v_wmma_f32_16x16x32_bf16 (CDNA5 wave32 WMMA).
// K-tile staging uses GLOBAL_LOAD_ASYNC_TO_LDS (ASYNCcnt) when available.
// ---------------------------------------------------------------------------

typedef __attribute__((ext_vector_type(16))) __bf16 v16bf;
typedef __attribute__((ext_vector_type(8)))  float  v8f;
typedef __attribute__((ext_vector_type(4)))  int    v4i;

union Frag16 {
    v16bf v;
    uint4 q[2];
};

union Pack16 {            // 16 bf16 <-> two 16B words (for packed LDS stores)
    __bf16 h[16];
    uint4  q[2];
};

__device__ __forceinline__ __bf16 f2bf(float f) { return (__bf16)f; }  // v_cvt_pk_bf16_f32

#if __has_builtin(__builtin_amdgcn_global_load_async_to_lds_b128) && \
    __has_builtin(__builtin_amdgcn_s_wait_asynccnt)
#define HAVE_ASYNC_LDS 1
#else
#define HAVE_ASYNC_LDS 0
#endif

#if HAVE_ASYNC_LDS
__device__ __forceinline__ void async_b128(const void* g, void* l) {
    __builtin_amdgcn_global_load_async_to_lds_b128(
        (__attribute__((address_space(1))) v4i*)(g),
        (__attribute__((address_space(3))) v4i*)(l),
        0, 0);
}
#endif

#define BT     16384   // B*T rows
#define CDIM   768
#define HDIM   64
#define QTILE  128     // query rows per block (kernel 2), rows per block (kernel 1)
#define KTILE  64      // key rows per flash iteration
#define KCHUNK 32      // WMMA K depth for bf16

// ---------------------------------------------------------------------------
// Kernel 1: out[r, 0:64] = x[r, 0:768] @ W  (bf16 output), 3 weight matrices
// grid = (BT/128, 3), block = 256 (8 waves). blockIdx.y: 0->Q, 1->K, 2->V.
// Q output is pre-scaled by 768^-0.5 so the flash kernel skips score scaling.
// ---------------------------------------------------------------------------
__global__ void __launch_bounds__(256)
qkv_proj_kernel(const float* __restrict__ x,
                const float* __restrict__ Wk,
                const float* __restrict__ Wq,
                const float* __restrict__ Wv,
                __bf16* __restrict__ Qb,
                __bf16* __restrict__ Kb,
                __bf16* __restrict__ Vb)
{
    const int which = blockIdx.y;
    const float* __restrict__ W = (which == 0) ? Wq : (which == 1) ? Wk : Wv;
    __bf16* __restrict__ Out = (which == 0) ? Qb : (which == 1) ? Kb : Vb;
    const float outscale = (which == 0) ? 0.03608439182435161f : 1.0f;  // 768^-0.5

    const int row0 = blockIdx.x * QTILE;
    const int tid  = threadIdx.x;
    const int wave = tid >> 5;
    const int lane = tid & 31;
    const int l16  = lane & 15;
    const bool hi  = (lane >= 16);

    __shared__ alignas(16) __bf16 xt[QTILE * KCHUNK];  // 8 KB
    __shared__ alignas(16) __bf16 wt[HDIM * KCHUNK];   // 4 KB, transposed [n][k]

    v8f acc[4];
#pragma unroll
    for (int n = 0; n < 4; ++n) acc[n] = {};

    for (int c0 = 0; c0 < CDIM; c0 += KCHUNK) {
        __syncthreads();
        // --- stage x tile (128x32 f32 -> bf16), packed 16B LDS stores ---
        {
            const int r  = tid >> 1;
            const int cb = (tid & 1) * 16;
            const float* src = x + (size_t)(row0 + r) * CDIM + c0 + cb;
            Pack16 p;
#pragma unroll
            for (int i = 0; i < 16; ++i) p.h[i] = f2bf(src[i]);
            *(uint4*)&xt[r * KCHUNK + cb]     = p.q[0];
            *(uint4*)&xt[r * KCHUNK + cb + 8] = p.q[1];
        }
        // --- stage W tile transposed (32x64 f32 -> wt[n][k] bf16) ---
        {
#pragma unroll
            for (int i = 0; i < 8; ++i) {
                const int e = tid * 8 + i;
                const int k = e >> 6;
                const int n = e & 63;
                wt[n * KCHUNK + k] = f2bf(W[(size_t)(c0 + k) * HDIM + n]);
            }
        }
        __syncthreads();

        // --- A fragment: 16x32 bf16, rows = wave's 16 rows ---
        Frag16 a;
        {
            const int r  = wave * 16 + l16;
            const int k0 = hi ? 8 : 0;   // lanes 0-15: K 0..7,16..23 ; 16-31: 8..15,24..31
            a.q[0] = *(const uint4*)&xt[r * KCHUNK + k0];
            a.q[1] = *(const uint4*)&xt[r * KCHUNK + k0 + 16];
        }
        // --- 4 B fragments (32x16 each) and WMMAs ---
#pragma unroll
        for (int n = 0; n < 4; ++n) {
            Frag16 bfr;
            const int col = n * 16 + l16;
            const int k0  = hi ? 8 : 0;   // lanes 0-15: K 0..7,16..23 ; 16-31: 8..15,24..31
            bfr.q[0] = *(const uint4*)&wt[col * KCHUNK + k0];
            bfr.q[1] = *(const uint4*)&wt[col * KCHUNK + k0 + 16];
            acc[n] = __builtin_amdgcn_wmma_f32_16x16x32_bf16(
                false, a.v, false, bfr.v, (short)0, acc[n], false, false);
        }
    }

    // --- store C layout -> bf16 workspace (Q pre-scaled) ---
    const int rbase = row0 + wave * 16 + (hi ? 8 : 0);
#pragma unroll
    for (int n = 0; n < 4; ++n)
#pragma unroll
        for (int j = 0; j < 8; ++j)
            Out[(size_t)(rbase + j) * HDIM + n * 16 + l16] = f2bf(acc[n][j] * outscale);
}

// ---------------------------------------------------------------------------
// Kernel 2: flash attention with causal mask.
// grid = (T/128, B), block = 256 (8 waves, 16 query rows per wave).
// ---------------------------------------------------------------------------
__global__ void __launch_bounds__(256)
flash_attn_kernel(const __bf16* __restrict__ Qb,
                  const __bf16* __restrict__ Kb,
                  const __bf16* __restrict__ Vb,
                  float* __restrict__ out)
{
    const int b    = blockIdx.y;
    const int q0   = blockIdx.x * QTILE;
    const int tid  = threadIdx.x;
    const int wave = tid >> 5;
    const int lane = tid & 31;
    const int l16  = lane & 15;
    const bool hi  = (lane >= 16);
    const int base = b * 2048;                 // batch row offset in [B*T]

    __shared__ alignas(16) __bf16 Kt[KTILE * HDIM];    // [s][h]  8 KB
    __shared__ alignas(16) __bf16 Vt[HDIM * KTILE];    // [h][s]  8 KB (transposed)
    __shared__ alignas(16) __bf16 Pt[8][16 * KTILE];   // per-wave P staging 16 KB

    // --- Q fragments, resident in VGPRs for whole kernel (A layout, 2 K-chunks) ---
    Frag16 qA[2];
    {
        const int r = base + q0 + wave * 16 + l16;
#pragma unroll
        for (int c = 0; c < 2; ++c) {
            const int h0 = c * 32 + (hi ? 8 : 0);
            qA[c].q[0] = *(const uint4*)&Qb[(size_t)r * HDIM + h0];
            qA[c].q[1] = *(const uint4*)&Qb[(size_t)r * HDIM + h0 + 16];
        }
    }

    float m[8], l[8];
    v8f   o[4];
#pragma unroll
    for (int j = 0; j < 8; ++j) { m[j] = -__builtin_inff(); l[j] = 0.0f; }
#pragma unroll
    for (int n = 0; n < 4; ++n) o[n] = {};

    const int q0w   = q0 + wave * 16;
    const int ktmax = (q0 + QTILE - 1) >> 6;   // inclusive causal key-tile limit

    for (int kt = 0; kt <= ktmax; ++kt) {
        const int kb = kt * KTILE;
        __syncthreads();
        // --- stage K tile [s][h]: async global->LDS when available ---
        {
            const __bf16* src = Kb + (size_t)(base + kb) * HDIM + tid * 16;
#if HAVE_ASYNC_LDS
            async_b128(src,     &Kt[tid * 16]);
            async_b128(src + 8, &Kt[tid * 16 + 8]);
#else
            const uint4 a0 = *(const uint4*)(src);
            const uint4 a1 = *(const uint4*)(src + 8);
            *(uint4*)&Kt[tid * 16]     = a0;
            *(uint4*)&Kt[tid * 16 + 8] = a1;
#endif
        }
        // --- stage V tile transposed -> Vt[h][s] (manual: transpose in flight) ---
        {
            const int s  = tid >> 2;
            const int h0 = (tid & 3) * 16;
            const __bf16* src = Vb + (size_t)(base + kb + s) * HDIM + h0;
            Pack16 tmp;
            tmp.q[0] = *(const uint4*)(src);
            tmp.q[1] = *(const uint4*)(src + 8);
#pragma unroll
            for (int i = 0; i < 16; ++i) Vt[(h0 + i) * KTILE + s] = tmp.h[i];
        }
        if (kt < ktmax)  // pull next K tile toward L2/WGP$
            __builtin_prefetch(Kb + (size_t)(base + kb + KTILE) * HDIM, 0, 1);
#if HAVE_ASYNC_LDS
        __builtin_amdgcn_s_wait_asynccnt(0);
#endif
        __syncthreads();

        const bool active = (kb <= q0w + 15);  // wave-uniform: EXEC stays all-1s
        float alpha[8];
        if (active) {
            // ---- S = Q K^T (4 N-tiles x 2 K-chunks = 8 WMMAs); scale folded into Q ----
            v8f s[4];
#pragma unroll
            for (int n = 0; n < 4; ++n) {
                v8f c = {};
#pragma unroll
                for (int cc = 0; cc < 2; ++cc) {
                    Frag16 kB;
                    const int scol = n * 16 + l16;
                    const int h0   = cc * 32 + (hi ? 8 : 0);
                    kB.q[0] = *(const uint4*)&Kt[scol * HDIM + h0];
                    kB.q[1] = *(const uint4*)&Kt[scol * HDIM + h0 + 16];
                    c = __builtin_amdgcn_wmma_f32_16x16x32_bf16(
                        false, qA[cc].v, false, kB.v, (short)0, c, false, false);
                }
                s[n] = c;
            }
            // ---- causal mask (diagonal tiles only) ----
            const bool needmask = (kb + KTILE - 1 > q0w);
            if (needmask) {
#pragma unroll
                for (int n = 0; n < 4; ++n)
#pragma unroll
                    for (int j = 0; j < 8; ++j) {
                        const int qrow = q0w + j + (hi ? 8 : 0);
                        const int key  = kb + n * 16 + l16;
                        if (key > qrow) s[n][j] = -__builtin_inff();
                    }
            }
            // ---- online softmax: row stats across the 16-lane groups ----
#pragma unroll
            for (int j = 0; j < 8; ++j) {
                float rmax = fmaxf(fmaxf(s[0][j], s[1][j]), fmaxf(s[2][j], s[3][j]));
#pragma unroll
                for (int off = 1; off <= 8; off <<= 1)
                    rmax = fmaxf(rmax, __shfl_xor(rmax, off, 16));
                const float mn = fmaxf(m[j], rmax);
                alpha[j] = __expf(m[j] - mn);
                m[j] = mn;
                float rs = 0.0f;
#pragma unroll
                for (int n = 0; n < 4; ++n) {
                    const float e = __expf(s[n][j] - mn);
                    s[n][j] = e;
                    rs += e;
                }
#pragma unroll
                for (int off = 1; off <= 8; off <<= 1)
                    rs += __shfl_xor(rs, off, 16);
                l[j] = l[j] * alpha[j] + rs;
            }
            // ---- rescale O, stage P (C layout -> LDS, bf16) ----
#pragma unroll
            for (int n = 0; n < 4; ++n)
#pragma unroll
                for (int j = 0; j < 8; ++j) {
                    o[n][j] *= alpha[j];
                    Pt[wave][(j + (hi ? 8 : 0)) * KTILE + n * 16 + l16] = f2bf(s[n][j]);
                }
        }
        __syncthreads();  // make P visible / ordered before A-layout reload
        if (active) {
            // ---- reload P as A fragments, O += P V (8 WMMAs) ----
            Frag16 pA[2];
#pragma unroll
            for (int c = 0; c < 2; ++c) {
                const int s0 = c * 32 + (hi ? 8 : 0);
                pA[c].q[0] = *(const uint4*)&Pt[wave][l16 * KTILE + s0];
                pA[c].q[1] = *(const uint4*)&Pt[wave][l16 * KTILE + s0 + 16];
            }
#pragma unroll
            for (int n = 0; n < 4; ++n) {
#pragma unroll
                for (int c = 0; c < 2; ++c) {
                    Frag16 vB;
                    const int h  = n * 16 + l16;
                    const int s0 = c * 32 + (hi ? 8 : 0);
                    vB.q[0] = *(const uint4*)&Vt[h * KTILE + s0];
                    vB.q[1] = *(const uint4*)&Vt[h * KTILE + s0 + 16];
                    o[n] = __builtin_amdgcn_wmma_f32_16x16x32_bf16(
                        false, pA[c].v, false, vB.v, (short)0, o[n], false, false);
                }
            }
        }
    }

    // --- normalize (one reciprocal per row) and store fp32 output ---
    float linv[8];
#pragma unroll
    for (int j = 0; j < 8; ++j) linv[j] = __builtin_amdgcn_rcpf(l[j]);
    const int rbase = q0 + wave * 16 + (hi ? 8 : 0);
#pragma unroll
    for (int n = 0; n < 4; ++n)
#pragma unroll
        for (int j = 0; j < 8; ++j)
            out[(size_t)(base + rbase + j) * HDIM + n * 16 + l16] = o[n][j] * linv[j];
}

// ---------------------------------------------------------------------------
extern "C" void kernel_launch(void* const* d_in, const int* in_sizes, int n_in,
                              void* d_out, int out_size, void* d_ws, size_t ws_size,
                              hipStream_t stream)
{
    const float* x  = (const float*)d_in[0];
    const float* Wk = (const float*)d_in[1];
    const float* Wq = (const float*)d_in[2];
    const float* Wv = (const float*)d_in[3];

    // bf16 workspace: Q | K | V, each BT*HDIM elements (2 MB) -> 6 MB total
    __bf16* Qb = (__bf16*)d_ws;
    __bf16* Kb = Qb + (size_t)BT * HDIM;
    __bf16* Vb = Kb + (size_t)BT * HDIM;

    qkv_proj_kernel<<<dim3(BT / QTILE, 3), 256, 0, stream>>>(x, Wk, Wq, Wv, Qb, Kb, Vb);
    flash_attn_kernel<<<dim3(2048 / QTILE, 8), 256, 0, stream>>>(Qb, Kb, Vb, (float*)d_out);
}
